// CausalSelfAttention_51728586113284
// MI455X (gfx1250) — matrix operand order repaired and checked by reading, not compiled, not run on hardware
//
#include <hip/hip_runtime.h>

// ---------------- problem constants ----------------
constexpr int Bn = 2;
constexpr int Tn = 2048;
constexpr int Cn = 2048;
constexpr int Hn = 16;
constexpr int Dn = 128;
constexpr int BT = Bn * Tn;          // 4096
constexpr int C3 = 3 * Cn;           // 6144

// ---------------- CDNA5 WMMA / TDM types ----------------
typedef __attribute__((ext_vector_type(16))) __bf16 bf16x16;
typedef __attribute__((ext_vector_type(8)))  float  f32x8;
typedef __attribute__((ext_vector_type(4)))  unsigned int u32x4;
typedef __attribute__((ext_vector_type(8)))  int i32x8;
typedef __attribute__((ext_vector_type(4)))  int i32x4;

union FragU { bf16x16 v; uint4 u[2]; };

#if defined(__has_builtin)
#if __has_builtin(__builtin_amdgcn_tensor_load_to_lds)
#define HAVE_TDM 1
#endif
#endif
#ifndef HAVE_TDM
#define HAVE_TDM 0
#endif

__device__ __forceinline__ void wait_tensorcnt0() {
#if defined(__has_builtin) && __has_builtin(__builtin_amdgcn_s_wait_tensorcnt)
    __builtin_amdgcn_s_wait_tensorcnt(0);
#else
    asm volatile("s_wait_tensorcnt 0x0" ::: "memory");
#endif
}
__device__ __forceinline__ void wait_tensorcnt1() {
#if defined(__has_builtin) && __has_builtin(__builtin_amdgcn_s_wait_tensorcnt)
    __builtin_amdgcn_s_wait_tensorcnt(1);
#else
    asm volatile("s_wait_tensorcnt 0x1" ::: "memory");
#endif
}

#if HAVE_TDM
__device__ __forceinline__ void tdm_issue(u32x4 g0, i32x8 g1) {
    i32x4 z4 = {};
#if defined(__clang_major__) && (__clang_major__ >= 23)
    i32x8 z8 = {};
    __builtin_amdgcn_tensor_load_to_lds(g0, g1, z4, z4, z8, 0);
#else
    __builtin_amdgcn_tensor_load_to_lds(g0, g1, z4, z4, 0);
#endif
}

// DMA a 2D bf16 tile [tile_n rows x tile_k elems], row stride = stride elems,
// from global gsrc into LDS at ldst. D# layout per CDNA5 ISA.
__device__ __forceinline__ void tdm_load_tile_2d(const unsigned short* gsrc,
                                                 const unsigned short* ldst,
                                                 int tile_k, int tile_n,
                                                 long stride_elems,
                                                 int tensor_k, int tensor_n) {
    const unsigned long long ga = (unsigned long long)(const void*)gsrc;
    const unsigned lds = (unsigned)(unsigned long long)(const void*)ldst;
    u32x4 g0;
    g0.x = 1u;                                          // count=1 (valid user D#)
    g0.y = lds;                                         // lds_addr [63:32]
    g0.z = (unsigned)ga;                                // global_addr [95:64]
    g0.w = (unsigned)((ga >> 32) & 0x01FFFFFFu) | (2u << 30);  // addr[56:32] | type=2
    const unsigned long long s0 = (unsigned long long)stride_elems;
    i32x8 g1;
    g1[0] = (int)(1u << 16);                            // data_size=1 (2 bytes)
    g1[1] = (int)(((unsigned)tensor_k & 0xffffu) << 16);          // tensor_dim0 lo16
    g1[2] = (int)((((unsigned)tensor_k >> 16) & 0xffffu)
                  | (((unsigned)tensor_n & 0xffffu) << 16));      // dim0 hi | dim1 lo
    g1[3] = (int)((((unsigned)tensor_n >> 16) & 0xffffu)
                  | (((unsigned)tile_k & 0xffffu) << 16));        // dim1 hi | tile_dim0
    g1[4] = (int)((unsigned)tile_n & 0xffffu);                    // tile_dim1 (dim2=0)
    g1[5] = (int)(unsigned)(s0 & 0xffffffffu);                    // dim0_stride[31:0]
    g1[6] = (int)(unsigned)((s0 >> 32) & 0xffffu);                // dim0_stride[47:32]
    g1[7] = 0;
    tdm_issue(g0, g1);
}
#endif  // HAVE_TDM

// float -> bf16 (round to nearest even)
__device__ __forceinline__ unsigned short f2bf(float x) {
    unsigned u = __float_as_uint(x);
    u = (u + 0x7FFFu + ((u >> 16) & 1u)) >> 16;
    return (unsigned short)u;
}
__device__ __forceinline__ float bf2f(unsigned short s) {
    return __uint_as_float(((unsigned)s) << 16);
}

// A-fragment (16x32 bf16): lanes 0-15 hold K 0..15, 16-31 hold K 16..31
__device__ __forceinline__ bf16x16 load_a_frag(const unsigned short* base,
                                               long ld, int row, int k0, int lane) {
    FragU f;
    const unsigned short* p = base + (long)row * ld + k0 + ((lane >> 4) & 1) * 16;
    f.u[0] = *(const uint4*)(p);
    f.u[1] = *(const uint4*)(p + 8);
    return f.v;
}
// B-fragment from 16 contiguous bf16 (lanes 0-15: K 0..15, lanes 16-31: K 16..31)
__device__ __forceinline__ bf16x16 load_b_contig(const unsigned short* p) {
    FragU f;
    f.u[0] = *(const uint4*)(p);
    f.u[1] = *(const uint4*)(p + 8);
    return f.v;
}

// ---------------- fp32 -> bf16 conversion ----------------
__global__ void cvt_f32_to_bf16(const float* __restrict__ src,
                                unsigned short* __restrict__ dst, long n4) {
    long i = (long)blockIdx.x * blockDim.x + threadIdx.x;
    if (i >= n4) return;
    const float4 s = ((const float4*)src)[i];
    ushort4 d;
    d.x = f2bf(s.x); d.y = f2bf(s.y); d.z = f2bf(s.z); d.w = f2bf(s.w);
    ((ushort4*)dst)[i] = d;
}

// ---------------- bf16 GEMM: out[m][n] = sum_k A[m][k] * Bm[n][k] ----------------
// block = 8 waves -> 128M x 64N; B tile (64x32) staged into LDS via TDM, dbl-buffered.
template <bool OUT_F32>
__device__ __forceinline__ void gemm_dev(const unsigned short* __restrict__ A,
                                         const unsigned short* __restrict__ Bm,
                                         unsigned short* __restrict__ outb,
                                         float* __restrict__ outf,
                                         int M, int N, int K) {
    __shared__ unsigned short bt[2][64][32];   // 2 x 4 KB

    const int lane = threadIdx.x & 31;
    const int wv = threadIdx.x >> 5;
    const int nblocks = N >> 6;
    const int mblk = blockIdx.x / nblocks;
    const int nblk = blockIdx.x % nblocks;
    const int m0 = mblk * 128 + wv * 16;
    const int n0 = nblk * 64;
    const int arow = m0 + (lane & 15);
    const int half = (lane >> 4) & 1;
    const int col = lane & 15;
    const int niter = K >> 5;

    f32x8 acc[4] = {};

#if HAVE_TDM
    if (wv == 0)   // one wave issues the DMA for the whole block
        tdm_load_tile_2d(Bm + (long)n0 * K, &bt[0][0][0], 32, 64, K, K, N - n0);
    for (int it = 0; it < niter; ++it) {
        const int k0 = it << 5;
        if (wv == 0) {
            if (it + 1 < niter) {
                tdm_load_tile_2d(Bm + (long)n0 * K + (k0 + 32),
                                 &bt[(it + 1) & 1][0][0], 32, 64, K, K, N - n0);
                wait_tensorcnt1();   // oldest (current buffer) complete
            } else {
                wait_tensorcnt0();
            }
        }
        __syncthreads();
        const unsigned short(*cur)[32] = bt[it & 1];
        if (k0 + 32 < K)
            __builtin_prefetch(A + (long)arow * K + k0 + 32, 0, 0);
        bf16x16 a = load_a_frag(A, K, arow, k0, lane);
#pragma unroll
        for (int i = 0; i < 4; ++i) {
            bf16x16 b = load_b_contig(&cur[i * 16 + col][half * 16]);
            acc[i] = __builtin_amdgcn_wmma_f32_16x16x32_bf16(
                false, a, false, b, (short)0, acc[i], false, false);
        }
        __syncthreads();  // all reads of this buffer done before TDM reuses it
    }
#else
    for (int it = 0; it < niter; ++it) {
        const int k0 = it << 5;
        __syncthreads();
        {   // cooperative stage: 256 threads x 16B = 4 KB tile
            const int nl = threadIdx.x >> 2;
            const int ko = (threadIdx.x & 3) * 8;
            *(uint4*)&bt[0][nl][ko] = *(const uint4*)(Bm + (long)(n0 + nl) * K + k0 + ko);
        }
        __syncthreads();
        bf16x16 a = load_a_frag(A, K, arow, k0, lane);
#pragma unroll
        for (int i = 0; i < 4; ++i) {
            bf16x16 b = load_b_contig(&bt[0][i * 16 + col][half * 16]);
            acc[i] = __builtin_amdgcn_wmma_f32_16x16x32_bf16(
                false, a, false, b, (short)0, acc[i], false, false);
        }
    }
#endif

    // C layout: element j -> row m0 + j + 8*half, col n0 + (lane&15)
#pragma unroll
    for (int i = 0; i < 4; ++i)
#pragma unroll
        for (int j = 0; j < 8; ++j) {
            const long m = m0 + j + half * 8;
            const long n = n0 + i * 16 + col;
            if (OUT_F32) outf[m * N + n] = acc[i][j];
            else         outb[m * N + n] = f2bf(acc[i][j]);
        }
}

__global__ __launch_bounds__(256)
void gemm_bf16_to_bf16(const unsigned short* A, const unsigned short* Bm,
                       unsigned short* outb, int M, int N, int K) {
    gemm_dev<false>(A, Bm, outb, nullptr, M, N, K);
}
__global__ __launch_bounds__(256)
void gemm_bf16_to_f32(const unsigned short* A, const unsigned short* Bm,
                      float* outf, int M, int N, int K) {
    gemm_dev<true>(A, Bm, nullptr, outf, M, N, K);
}

// ---------------- RoPE (in place on bf16 qkv, sel 0 = q, sel 1 = k) ----------------
__global__ void rope_kernel(unsigned short* __restrict__ qkv,
                            const float* __restrict__ cosb,
                            const float* __restrict__ sinb) {
    long idx = (long)blockIdx.x * blockDim.x + threadIdx.x;
    const int D2 = Dn / 2;
    int i = (int)(idx % D2);  idx /= D2;
    int h = (int)(idx % Hn);  idx /= Hn;
    int sel = (int)(idx % 2); idx /= 2;
    int t = (int)(idx % Tn);
    int b = (int)(idx / Tn);
    if (b >= Bn) return;
    const float c = cosb[t * D2 + i];
    const float s = sinb[t * D2 + i];
    unsigned short* p = qkv + ((long)(b * Tn + t)) * C3 + sel * Cn + h * Dn + 2 * i;
    const float x0 = bf2f(p[0]);
    const float x1 = bf2f(p[1]);
    p[0] = f2bf(x0 * c - x1 * s);
    p[1] = f2bf(x0 * s + x1 * c);
}

// ---------------- flash attention: one wave per (b, h, 16-row q tile) ----------------
__global__ __launch_bounds__(32)
void flash_attn(const unsigned short* __restrict__ qkv,
                unsigned short* __restrict__ obuf) {
    __shared__ unsigned short lds_kt[32][Dn];  // K tile [kv][d], TDM target, 8 KB
    __shared__ unsigned short lds_vt[Dn][32];  // V^T tile [d][kv], 8 KB
    __shared__ unsigned short lds_p[16][32];   // P tile [m][kv], 1 KB

    const int lane = threadIdx.x;
    const int qtiles = Tn / 16;
    int blk = blockIdx.x;
    const int qt = blk % qtiles;  blk /= qtiles;
    const int h = blk % Hn;
    const int b = blk / Hn;
    const int q0 = qt * 16;
    const int half = (lane >> 4) & 1;
    const int col = lane & 15;

    const unsigned short* qbase = qkv + (long)b * Tn * C3 + h * Dn;
    const unsigned short* kbase = qbase + Cn;
    const unsigned short* vbase = qbase + 2 * Cn;

    // Q fragments: 16 rows x 128 d = 4 A-frags of 16x32
    bf16x16 qf[4];
    {
        const unsigned short* qrow = qbase + (long)(q0 + col) * C3 + half * 16;
#pragma unroll
        for (int kk = 0; kk < 4; ++kk) {
            FragU f;
            const unsigned short* p = qrow + kk * 32;
            f.u[0] = *(const uint4*)(p);
            f.u[1] = *(const uint4*)(p + 8);
            qf[kk] = f.v;
        }
    }

    float m_i[8], l_i[8];
    f32x8 oacc[8] = {};
#pragma unroll
    for (int j = 0; j < 8; ++j) { m_i[j] = -1e30f; l_i[j] = 0.0f; }

    const float scale = 0.08838834764831845f;  // 1/sqrt(128)
    const int kvend = q0 + 16;

    for (int kv0 = 0; kv0 < kvend; kv0 += 32) {
        // ---- DMA the K chunk (32 kv rows x 128 d) into LDS via TDM ----
#if HAVE_TDM
        tdm_load_tile_2d(kbase + (long)kv0 * C3, &lds_kt[0][0],
                         Dn, 32, C3, Dn, 32);
#else
#pragma unroll 4
        for (int j = 0; j < 32; ++j)
            *(uint2*)&lds_kt[j][lane * 4] =
                *(const uint2*)(kbase + (long)(kv0 + j) * C3 + lane * 4);
#endif

        // ---- stage V^T chunk into LDS (overlaps the TDM) ----
        __syncthreads();
#pragma unroll 4
        for (int j = 0; j < 32; ++j) {
            const unsigned short* vp = vbase + (long)(kv0 + j) * C3 + lane * 4;
            uint2 d4 = *(const uint2*)vp;
            lds_vt[lane * 4 + 0][j] = (unsigned short)(d4.x & 0xffff);
            lds_vt[lane * 4 + 1][j] = (unsigned short)(d4.x >> 16);
            lds_vt[lane * 4 + 2][j] = (unsigned short)(d4.y & 0xffff);
            lds_vt[lane * 4 + 3][j] = (unsigned short)(d4.y >> 16);
        }
#if HAVE_TDM
        wait_tensorcnt0();
#endif
        __syncthreads();

        // ---- S = Q K^T (two 16x16 tiles, K-frags from TDM-staged LDS) ----
        f32x8 s[2] = {};
#pragma unroll
        for (int nt2 = 0; nt2 < 2; ++nt2) {
#pragma unroll
            for (int kk = 0; kk < 4; ++kk) {
                bf16x16 bfr = load_b_contig(&lds_kt[nt2 * 16 + col][kk * 32 + half * 16]);
                s[nt2] = __builtin_amdgcn_wmma_f32_16x16x32_bf16(
                    false, qf[kk], false, bfr, (short)0, s[nt2], false, false);
            }
        }

        // ---- causal mask + scale + chunk row max ----
        float cmax[8];
#pragma unroll
        for (int j = 0; j < 8; ++j) {
            const int qm = q0 + j + half * 8;
            const int kn = kv0 + col;
            float v0 = s[0][j] * scale;  if (kn > qm)      v0 = -1e30f;
            float v1 = s[1][j] * scale;  if (kn + 16 > qm) v1 = -1e30f;
            s[0][j] = v0; s[1][j] = v1;
            float tv = fmaxf(v0, v1);
            tv = fmaxf(tv, __shfl_xor(tv, 1, 32));
            tv = fmaxf(tv, __shfl_xor(tv, 2, 32));
            tv = fmaxf(tv, __shfl_xor(tv, 4, 32));
            tv = fmaxf(tv, __shfl_xor(tv, 8, 32));
            cmax[j] = tv;
        }

        __syncthreads();  // previous P-frag reads finished before overwrite

        // ---- online softmax update; emit P in [m][kv] LDS layout ----
#pragma unroll
        for (int j = 0; j < 8; ++j) {
            const float mnew = fmaxf(m_i[j], cmax[j]);
            const float alpha = __expf(m_i[j] - mnew);
            const float p0 = __expf(s[0][j] - mnew);
            const float p1 = __expf(s[1][j] - mnew);
            float rs = p0 + p1;
            rs += __shfl_xor(rs, 1, 32);
            rs += __shfl_xor(rs, 2, 32);
            rs += __shfl_xor(rs, 4, 32);
            rs += __shfl_xor(rs, 8, 32);
            l_i[j] = l_i[j] * alpha + rs;
            m_i[j] = mnew;
#pragma unroll
            for (int dt = 0; dt < 8; ++dt) oacc[dt][j] *= alpha;
            const int pr = j + half * 8;
            lds_p[pr][col]      = f2bf(p0);
            lds_p[pr][col + 16] = f2bf(p1);
        }
        __syncthreads();

        // ---- P A-fragment (16x32 over kv) from LDS ----
        FragU pf;
        {
            const unsigned short* p = &lds_p[col][half * 16];
            pf.u[0] = *(const uint4*)(p);
            pf.u[1] = *(const uint4*)(p + 8);
        }

        // ---- O += P V : 8 d-tiles, B-frags from transposed V in LDS ----
#pragma unroll
        for (int dt = 0; dt < 8; ++dt) {
            bf16x16 vf = load_b_contig(&lds_vt[dt * 16 + col][half * 16]);
            oacc[dt] = __builtin_amdgcn_wmma_f32_16x16x32_bf16(
                false, pf.v, false, vf, (short)0, oacc[dt], false, false);
        }
    }

    // ---- normalize and store O (bf16, [B*T][C] with head offset) ----
#pragma unroll
    for (int dt = 0; dt < 8; ++dt)
#pragma unroll
        for (int j = 0; j < 8; ++j) {
            const long m = q0 + j + half * 8;
            const long n = h * Dn + dt * 16 + col;
            const float val = oacc[dt][j] / l_i[j];
            obuf[((long)b * Tn + m) * Cn + n] = f2bf(val);
        }
}

// ---------------- host-side launch ----------------
extern "C" void kernel_launch(void* const* d_in, const int* in_sizes, int n_in,
                              void* d_out, int out_size, void* d_ws, size_t ws_size,
                              hipStream_t stream) {
    const float* x      = (const float*)d_in[0];
    const float* w_qkv  = (const float*)d_in[1];
    const float* w_proj = (const float*)d_in[2];
    const float* fcos   = (const float*)d_in[3];
    const float* fsin   = (const float*)d_in[4];
    float* out = (float*)d_out;

    // bf16 workspace layout (~112 MB)
    unsigned short* xb     = (unsigned short*)d_ws;
    unsigned short* wqkvb  = xb     + (long)BT * Cn;
    unsigned short* wprojb = wqkvb  + (long)C3 * Cn;
    unsigned short* qkvb   = wprojb + (long)Cn * Cn;
    unsigned short* obuf   = qkvb   + (long)BT * C3;

    const int thr = 256;
    // 1) convert inputs to bf16 (vectorized x4)
    {
        long n4;
        n4 = (long)BT * Cn / 4;
        cvt_f32_to_bf16<<<(int)((n4 + thr - 1) / thr), thr, 0, stream>>>(x, xb, n4);
        n4 = (long)C3 * Cn / 4;
        cvt_f32_to_bf16<<<(int)((n4 + thr - 1) / thr), thr, 0, stream>>>(w_qkv, wqkvb, n4);
        n4 = (long)Cn * Cn / 4;
        cvt_f32_to_bf16<<<(int)((n4 + thr - 1) / thr), thr, 0, stream>>>(w_proj, wprojb, n4);
    }

    // 2) QKV GEMM: (BT x C) @ (3C x C)^T -> (BT x 3C) bf16
    {
        const int blocks = (BT / 128) * (C3 / 64);
        gemm_bf16_to_bf16<<<blocks, thr, 0, stream>>>(xb, wqkvb, qkvb, BT, C3, Cn);
    }

    // 3) RoPE on q and k slices
    {
        const long total = (long)Bn * Tn * 2 * Hn * (Dn / 2);
        rope_kernel<<<(int)(total / thr), thr, 0, stream>>>(qkvb, fcos, fsin);
    }

    // 4) causal flash attention -> obuf (BT x C) bf16
    {
        const int blocks = Bn * Hn * (Tn / 16);
        flash_attn<<<blocks, 32, 0, stream>>>(qkvb, obuf);
    }

    // 5) output projection: (BT x C) @ (C x C)^T -> f32 d_out
    {
        const int blocks = (BT / 128) * (Cn / 64);
        gemm_bf16_to_f32<<<blocks, thr, 0, stream>>>(obuf, wprojb, out, BT, Cn, Cn);
    }
}
